// TransformerBlock_66614942760983
// MI455X (gfx1250) — compile-verified, matrix operand order checked by reading, not run on hardware
//
#include <hip/hip_runtime.h>
#include <hip/hip_bf16.h>
#include <stdint.h>

#define B_   2
#define T_   2048
#define E_   1024
#define H_   16
#define HS_  64
#define FF_  4096
#define NTOK (B_*T_)

typedef __bf16 bf16;
typedef __attribute__((ext_vector_type(16))) __bf16 v16bf;
typedef __attribute__((ext_vector_type(8)))  __bf16 v8bf;
typedef __attribute__((ext_vector_type(8)))  float  v8f;

// 16-byte global -> LDS async copy (ASYNCcnt-tracked CDNA5 path).
__device__ __forceinline__ void cp_async16(void* lds, const void* gsrc) {
  unsigned loff = (unsigned)(size_t)(__attribute__((address_space(3))) char*)lds;
  asm volatile("global_load_async_to_lds_b128 %0, %1, off"
               :: "v"(loff), "v"(gsrc)
               : "memory");
}
__device__ __forceinline__ void wait_async() {
  asm volatile("s_wait_asynccnt 0x0" ::: "memory");
}

// 16-element bf16 fragment for WMMA A (16x32) / B (32x16) operands.
// lanes<16 hold K={0..7,16..23}, lanes>=16 hold K={8..15,24..31}
// -> two contiguous 16B LDS loads per lane.
__device__ __forceinline__ v16bf frag_ld(const bf16* rowPtr, int half) {
  const v8bf lo = *reinterpret_cast<const v8bf*>(rowPtr + 8 * half);
  const v8bf hi = *reinterpret_cast<const v8bf*>(rowPtr + 16 + 8 * half);
  v16bf r;
#pragma unroll
  for (int i = 0; i < 8; ++i) { r[i] = lo[i]; r[i + 8] = hi[i]; }
  return r;
}

// ---------------------------------------------------------------- casts
__global__ __launch_bounds__(256)
void cast_f32_bf16(const float* __restrict__ in, bf16* __restrict__ out, int n) {
  int i4 = (blockIdx.x * 256 + threadIdx.x) * 4;
  if (i4 < n) {
    float4 f = *reinterpret_cast<const float4*>(in + i4);
    out[i4 + 0] = (bf16)f.x;
    out[i4 + 1] = (bf16)f.y;
    out[i4 + 2] = (bf16)f.z;
    out[i4 + 3] = (bf16)f.w;
  }
}

// fp32 [R][C] -> bf16 [C][R] (per slice), 32x32 LDS tile transpose.
__global__ __launch_bounds__(256)
void transpose_cast(const float* __restrict__ in, bf16* __restrict__ out,
                    int R, int C) {
  __shared__ float t[32][33];
  const long slice = (long)blockIdx.z * R * C;
  const float* ip = in + slice;
  bf16* op = out + slice;
  const int r0 = blockIdx.y * 32, c0 = blockIdx.x * 32;
  const int tx = threadIdx.x & 31, ty = threadIdx.x >> 5;  // 32 x 8
#pragma unroll
  for (int i = 0; i < 32; i += 8)
    t[ty + i][tx] = ip[(long)(r0 + ty + i) * C + c0 + tx];
  __syncthreads();
#pragma unroll
  for (int i = 0; i < 32; i += 8)
    op[(long)(c0 + ty + i) * R + r0 + tx] = (bf16)t[tx][ty + i];
}

// ---------------------------------------------------------------- GEMM
// C[M,N] = A[M,K] * Bt[N,K]^T (+bias)(+relu)(*scale). Bt pre-transposed
// (N-major): both LDS tiles stage as contiguous 16B async copies.
// Double-buffered software pipeline, BK=64 (2 WMMA k-steps per stage).
// flags: bit0=relu, bit1=head [B,H,T,HS] out, bit2=head [B,H,HS,T] out,
//        bit3=scale output by E^-0.5 (folds attention score scaling into Q).
template <int BM, int BN, int WM, int WN>
__global__ __launch_bounds__((BM / WM) * (BN / WN) * 32)
void gemm_bf16_wmma(const bf16* __restrict__ A, int lda,
                    const bf16* __restrict__ Bt, int ldbt, long bHeadStride,
                    const float* __restrict__ bias, int biasHeadStride,
                    float* __restrict__ Cf, bf16* __restrict__ Cb, int ldc,
                    int K, int flags) {
  constexpr int WAVES_N = BN / WN;
  constexpr int NTHR = (BM / WM) * WAVES_N * 32;
  constexpr int SM = WM / 16, SN = WN / 16;
  constexpr int ACH = (BM * 8) / NTHR;  // 16B chunks/thread for A tile (BM x 64)
  constexpr int BCH = (BN * 8) / NTHR;

  __shared__ bf16 As[2][BM][64];
  __shared__ bf16 Bs[2][BN][64];

  const int tid  = threadIdx.x;
  const int wave = tid >> 5;
  const int lane = tid & 31;
  const int l16  = lane & 15;
  const int half = lane >> 4;

  const int rowBase = blockIdx.x * BM;
  const int colBase = blockIdx.y * BN;
  const int h       = blockIdx.z;
  const bf16* Bp = Bt + (long)h * bHeadStride;

  const int waveM = (wave / WAVES_N) * WM;
  const int waveN = (wave % WAVES_N) * WN;

  // staging chunk descriptors (8 chunks of 8 bf16 per 64-wide row)
  int aR[ACH], aC[ACH], bR[BCH], bC[BCH];
  const bf16* aSrc[ACH];
  const bf16* bSrc[BCH];
#pragma unroll
  for (int i = 0; i < ACH; ++i) {
    int ch = tid + i * NTHR;
    aR[i] = ch >> 3; aC[i] = (ch & 7) * 8;
    aSrc[i] = A + (long)(rowBase + aR[i]) * lda + aC[i];
  }
#pragma unroll
  for (int i = 0; i < BCH; ++i) {
    int ch = tid + i * NTHR;
    bR[i] = ch >> 3; bC[i] = (ch & 7) * 8;
    bSrc[i] = Bp + (long)(colBase + bR[i]) * ldbt + bC[i];
  }

  auto stage = [&](int buf) {
#pragma unroll
    for (int i = 0; i < ACH; ++i) {
      cp_async16(&As[buf][aR[i]][aC[i]], aSrc[i]);
      aSrc[i] += 64;
    }
#pragma unroll
    for (int i = 0; i < BCH; ++i) {
      cp_async16(&Bs[buf][bR[i]][bC[i]], bSrc[i]);
      bSrc[i] += 64;
    }
  };

  v8f acc[SM][SN];
#pragma unroll
  for (int a = 0; a < SM; ++a)
#pragma unroll
    for (int b = 0; b < SN; ++b)
#pragma unroll
      for (int r = 0; r < 8; ++r) acc[a][b][r] = 0.0f;

  stage(0);
  int buf = 0;
  for (int k0 = 0; k0 < K; k0 += 64) {
    wait_async();
    __syncthreads();
    if (k0 + 64 < K) stage(buf ^ 1);

#pragma unroll
    for (int ks = 0; ks < 2; ++ks) {
      v16bf af[SM], bfr[SN];
#pragma unroll
      for (int a = 0; a < SM; ++a)
        af[a] = frag_ld(&As[buf][waveM + a * 16 + l16][ks * 32], half);
#pragma unroll
      for (int b = 0; b < SN; ++b)
        bfr[b] = frag_ld(&Bs[buf][waveN + b * 16 + l16][ks * 32], half);
#pragma unroll
      for (int a = 0; a < SM; ++a)
#pragma unroll
        for (int b = 0; b < SN; ++b)
          acc[a][b] = __builtin_amdgcn_wmma_f32_16x16x32_bf16(
              false, af[a], false, bfr[b], (short)0, acc[a][b], false, false);
    }
    buf ^= 1;
  }

  const int doRelu   = flags & 1;
  const int headOut  = flags & 2;
  const int headOutT = flags & 4;
  const int doScale  = flags & 8;
#pragma unroll
  for (int a = 0; a < SM; ++a) {
#pragma unroll
    for (int b = 0; b < SN; ++b) {
#pragma unroll
      for (int r = 0; r < 8; ++r) {
        int row = rowBase + waveM + a * 16 + r + 8 * half;
        int nl  = waveN + b * 16 + l16;
        int col = colBase + nl;
        float v = acc[a][b][r];
        if (bias) v += bias[(long)h * biasHeadStride + ((flags & 6) ? nl : col)];
        if (doRelu) v = v > 0.0f ? v : 0.0f;
        if (doScale) v *= 0.03125f;  // E^-0.5, reference scales by embed dim
        int bb = row >> 11;  // row / T_
        int t  = row & (T_ - 1);
        if (headOut) {
          Cb[(((long)bb * H_ + h) * T_ + t) * HS_ + nl] = (bf16)v;
        } else if (headOutT) {
          Cb[(((long)bb * H_ + h) * HS_ + nl) * T_ + t] = (bf16)v;
        } else {
          long idx = (long)row * ldc + col;
          if (Cf) Cf[idx] = v;
          if (Cb) Cb[idx] = (bf16)v;
        }
      }
    }
  }
}

// ---------------------------------------------------------------- attention
// Flash-style. q-tile = 128 rows (8 waves x 16), key-tile = 64, async
// double-buffered K/V staging. V arrives pre-transposed [B,H,HS,T]; Q is
// pre-scaled by E^-0.5 at projection time. grid = (T/128, H, B), block = 256.
__global__ __launch_bounds__(256)
void attn_wmma(const bf16* __restrict__ Q, const bf16* __restrict__ Kv,
               const bf16* __restrict__ Vt, bf16* __restrict__ Out, int causal) {
  __shared__ bf16 Ks[2][64][64];   // [key][hs]
  __shared__ bf16 VTs[2][64][64];  // [hs][key]
  __shared__ bf16 Ps[8][16][64];   // per-wave P tile [qrow][key]

  const int tid  = threadIdx.x;
  const int wave = tid >> 5;
  const int lane = tid & 31;
  const int l16  = lane & 15;
  const int half = lane >> 4;

  const int qt = blockIdx.x, h = blockIdx.y, b = blockIdx.z;
  const int q0 = qt * 128 + wave * 16;
  const long bh   = ((long)b * H_ + h) * T_;   // rows of Q/K
  const long bhHS = ((long)b * H_ + h) * HS_;  // rows of Vt

  v16bf qf[2];
#pragma unroll
  for (int ks = 0; ks < 2; ++ks)
    qf[ks] = frag_ld(&Q[(bh + q0 + l16) * HS_ + ks * 32], half);

  // K tile 64x64 and V^T tile 64x64: 512 chunks each, 2/thread
  const int ch0 = tid, ch1 = tid + 256;
  const int kk0 = ch0 >> 3, kc0 = (ch0 & 7) * 8;
  const int kk1 = ch1 >> 3, kc1 = (ch1 & 7) * 8;
  const bf16* kSrc0 = Kv + (bh + kk0) * HS_ + kc0;
  const bf16* kSrc1 = Kv + (bh + kk1) * HS_ + kc1;
  const bf16* vSrc0 = Vt + (bhHS + kk0) * T_ + kc0;
  const bf16* vSrc1 = Vt + (bhHS + kk1) * T_ + kc1;

  auto stage = [&](int buf) {
    cp_async16(&Ks[buf][kk0][kc0], kSrc0);
    cp_async16(&Ks[buf][kk1][kc1], kSrc1);
    cp_async16(&VTs[buf][kk0][kc0], vSrc0);
    cp_async16(&VTs[buf][kk1][kc1], vSrc1);
    kSrc0 += 64 * HS_; kSrc1 += 64 * HS_;  // next 64 keys (K rows)
    vSrc0 += 64;       vSrc1 += 64;        // next 64 keys (Vt cols)
  };

  v8f o[4];
  float rmax[8], rsum[8];
#pragma unroll
  for (int t = 0; t < 4; ++t)
#pragma unroll
    for (int r = 0; r < 8; ++r) o[t][r] = 0.0f;
#pragma unroll
  for (int r = 0; r < 8; ++r) { rmax[r] = -3.0e38f; rsum[r] = 0.0f; }

  const int nj = causal ? (2 * qt + 2) : (T_ / 64);
  stage(0);
  int buf = 0;
  for (int j = 0; j < nj; ++j) {
    wait_async();
    __syncthreads();
    if (j + 1 < nj) stage(buf ^ 1);

    // wave-uniform causal classification of this 64-key tile vs 16-row strip
    const bool fullyMasked = causal && (j * 64 > q0 + 15);   // skip all compute
    const bool diagTile    = causal && (j * 64 + 63 > q0);   // per-elem mask

    if (!fullyMasked) {
      // S = Q * K^T  (16 q-rows x 64 keys); Q carries the E^-0.5 scale
      v8f s[4];
#pragma unroll
      for (int nt = 0; nt < 4; ++nt) {
#pragma unroll
        for (int r = 0; r < 8; ++r) s[nt][r] = 0.0f;
#pragma unroll
        for (int ks = 0; ks < 2; ++ks) {
          v16bf kf = frag_ld(&Ks[buf][nt * 16 + l16][ks * 32], half);
          s[nt] = __builtin_amdgcn_wmma_f32_16x16x32_bf16(
              false, qf[ks], false, kf, (short)0, s[nt], false, false);
        }
      }

      if (diagTile) {
#pragma unroll
        for (int nt = 0; nt < 4; ++nt) {
          int key = j * 64 + nt * 16 + l16;
#pragma unroll
          for (int r = 0; r < 8; ++r) {
            int qrow = q0 + r + 8 * half;
            if (key > qrow) s[nt][r] = -3.0e38f;
          }
        }
      }

      // online softmax (row = r + 8*half; 16-lane row reductions in wave32)
#pragma unroll
      for (int r = 0; r < 8; ++r) {
        float m = fmaxf(fmaxf(s[0][r], s[1][r]), fmaxf(s[2][r], s[3][r]));
#pragma unroll
        for (int off = 8; off >= 1; off >>= 1) m = fmaxf(m, __shfl_xor(m, off, 32));
        float nm = fmaxf(rmax[r], m);
        float scl = __expf(rmax[r] - nm);
        float rs = 0.0f;
#pragma unroll
        for (int nt = 0; nt < 4; ++nt) {
          float p = __expf(s[nt][r] - nm);
          s[nt][r] = p;
          rs += p;
        }
#pragma unroll
        for (int off = 8; off >= 1; off >>= 1) rs += __shfl_xor(rs, off, 32);
        rsum[r] = rsum[r] * scl + rs;
        rmax[r] = nm;
#pragma unroll
        for (int ont = 0; ont < 4; ++ont) o[ont][r] *= scl;
      }

      // P (C-layout) -> per-wave LDS -> A-fragment layout (same-wave, in-order)
#pragma unroll
      for (int nt = 0; nt < 4; ++nt)
#pragma unroll
        for (int r = 0; r < 8; ++r)
          Ps[wave][r + 8 * half][nt * 16 + l16] = (bf16)s[nt][r];
      asm volatile("s_wait_dscnt 0" ::: "memory");

      v16bf pf[2];
#pragma unroll
      for (int ks = 0; ks < 2; ++ks)
        pf[ks] = frag_ld(&Ps[wave][l16][ks * 32], half);
#pragma unroll
      for (int ont = 0; ont < 4; ++ont)
#pragma unroll
        for (int ks = 0; ks < 2; ++ks) {
          v16bf vf = frag_ld(&VTs[buf][ont * 16 + l16][ks * 32], half);
          o[ont] = __builtin_amdgcn_wmma_f32_16x16x32_bf16(
              false, pf[ks], false, vf, (short)0, o[ont], false, false);
        }
    }
    buf ^= 1;
  }

  // finalize, write heads-concat layout [b, t, h*HS + n]
#pragma unroll
  for (int ont = 0; ont < 4; ++ont)
#pragma unroll
    for (int r = 0; r < 8; ++r) {
      int t = q0 + r + 8 * half;
      float v = o[ont][r] / rsum[r];
      Out[((long)b * T_ + t) * E_ + h * HS_ + ont * 16 + l16] = (bf16)v;
    }
}

// ---------------------------------------------------------------- layernorm
__global__ __launch_bounds__(256)
void ln_fused(const float* __restrict__ acc, const float* __restrict__ bias,
              const float* __restrict__ resid, const float* __restrict__ g,
              const float* __restrict__ beta, float* __restrict__ outF,
              bf16* __restrict__ outB) {
  __shared__ float red[8];
  const long base = (long)blockIdx.x * E_;
  const int tid = threadIdx.x;

  float v[4];
  float s = 0.0f;
#pragma unroll
  for (int i = 0; i < 4; ++i) {
    int e = tid + i * 256;
    v[i] = acc[base + e] + bias[e] + resid[base + e];
    s += v[i];
  }
#pragma unroll
  for (int off = 16; off >= 1; off >>= 1) s += __shfl_xor(s, off, 32);
  if ((tid & 31) == 0) red[tid >> 5] = s;
  __syncthreads();
  float tot = 0.0f;
#pragma unroll
  for (int i = 0; i < 8; ++i) tot += red[i];
  const float mean = tot * (1.0f / E_);
  __syncthreads();

  float vs = 0.0f;
#pragma unroll
  for (int i = 0; i < 4; ++i) { float d = v[i] - mean; vs += d * d; }
#pragma unroll
  for (int off = 16; off >= 1; off >>= 1) vs += __shfl_xor(vs, off, 32);
  if ((tid & 31) == 0) red[tid >> 5] = vs;
  __syncthreads();
  float vtot = 0.0f;
#pragma unroll
  for (int i = 0; i < 8; ++i) vtot += red[i];
  const float rs = rsqrtf(vtot * (1.0f / E_) + 1e-5f);

#pragma unroll
  for (int i = 0; i < 4; ++i) {
    int e = tid + i * 256;
    float y = (v[i] - mean) * rs * g[e] + beta[e];
    if (outF) outF[base + e] = y;
    if (outB) outB[base + e] = (bf16)y;
  }
}

// ---------------------------------------------------------------- launcher
extern "C" void kernel_launch(void* const* d_in, const int* in_sizes, int n_in,
                              void* d_out, int out_size, void* d_ws, size_t ws_size,
                              hipStream_t stream) {
  (void)in_sizes; (void)n_in; (void)out_size; (void)ws_size;

  char* base = (char*)d_ws;
  size_t off = 0;
  auto nxt = [&](size_t bytes) -> void* {
    void* p = base + off;
    off = (off + bytes + 255) & ~(size_t)255;
    return p;
  };
  constexpr size_t SZ_X   = (size_t)NTOK * E_ * 2;
  constexpr size_t SZ_WEE = (size_t)E_ * E_ * 2;
  constexpr size_t SZ_WEF = (size_t)E_ * FF_ * 2;
  constexpr size_t SZ_QKV = (size_t)B_ * H_ * T_ * HS_ * 2;
  constexpr size_t SZ_F   = (size_t)NTOK * E_ * 4;
  constexpr size_t SZ_HID = (size_t)NTOK * FF_ * 2;

  bf16* Xsa  = (bf16*)nxt(SZ_X);
  bf16* Xc   = (bf16*)nxt(SZ_X);
  bf16* saWq = (bf16*)nxt(SZ_WEE);   // all weights stored TRANSPOSED [N][K]
  bf16* saWk = (bf16*)nxt(SZ_WEE);
  bf16* saWv = (bf16*)nxt(SZ_WEE);
  bf16* saWp = (bf16*)nxt(SZ_WEE);
  bf16* caWq = (bf16*)nxt(SZ_WEE);
  bf16* caWk = (bf16*)nxt(SZ_WEE);
  bf16* caWv = (bf16*)nxt(SZ_WEE);
  bf16* caWp = (bf16*)nxt(SZ_WEE);
  bf16* W1   = (bf16*)nxt(SZ_WEF);
  bf16* W2   = (bf16*)nxt(SZ_WEF);
  bf16* Qb   = (bf16*)nxt(SZ_QKV);
  bf16* Kb   = (bf16*)nxt(SZ_QKV);
  bf16* Vb   = (bf16*)nxt(SZ_QKV);   // [B,H,HS,T]
  bf16* attnO= (bf16*)nxt(SZ_X);
  float* proj= (float*)nxt(SZ_F);
  float* x1f = (float*)nxt(SZ_F);
  bf16* x1b  = (bf16*)nxt(SZ_X);
  float* x2f = (float*)nxt(SZ_F);
  bf16* x2b  = (bf16*)nxt(SZ_X);
  bf16* hid  = (bf16*)nxt(SZ_HID);
  float* ffo = (float*)nxt(SZ_F);

  auto inF = [&](int i) { return (const float*)d_in[i]; };
  auto cast = [&](int i, bf16* dst, int n) {
    cast_f32_bf16<<<dim3((n / 4 + 255) / 256), 256, 0, stream>>>(inF(i), dst, n);
  };
  auto tcast = [&](int i, bf16* dst, int R, int C, int S) {
    transpose_cast<<<dim3(C / 32, R / 32, S), 256, 0, stream>>>(inF(i), dst, R, C);
  };
  // big flat GEMM: 128x128 block tile, 256 threads
  auto gemmB = [&](const bf16* A, int lda, const bf16* Bt, int ldbt,
                   const float* bias, float* Cf, bf16* Cb, int ldc,
                   int M, int N, int K, int flags) {
    gemm_bf16_wmma<128, 128, 64, 32><<<dim3(M / 128, N / 128, 1), 256, 0, stream>>>(
        A, lda, Bt, ldbt, 0, bias, 0, Cf, Cb, ldc, K, flags);
  };
  // per-head projection GEMM: 64x64 block tile, 128 threads, grid.z = H
  auto gemmH = [&](const bf16* A, const bf16* Bt, const float* bias,
                   bf16* Cb, int flags) {
    gemm_bf16_wmma<64, 64, 32, 32><<<dim3(NTOK / 64, 1, H_), 128, 0, stream>>>(
        A, E_, Bt, E_, (long)HS_ * E_, bias, HS_, nullptr, Cb, 0, E_, flags);
  };

  // ---- one-time bf16 staging: activations flat, weights transposed
  cast(0, Xsa, NTOK * E_);
  cast(1, Xc, NTOK * E_);
  tcast(4,  saWq, E_, HS_, H_);  tcast(6,  saWk, E_, HS_, H_);
  tcast(8,  saWv, E_, HS_, H_);  tcast(10, saWp, E_, E_, 1);
  tcast(12, caWq, E_, HS_, H_);  tcast(14, caWk, E_, HS_, H_);
  tcast(16, caWv, E_, HS_, H_);  tcast(18, caWp, E_, E_, 1);
  tcast(20, W1, E_, FF_, 1);     tcast(22, W2, FF_, E_, 1);

  // ---- self-attention (causal); Q pre-scaled by E^-0.5 (flag bit3)
  gemmH(Xsa, saWq, inF(5), Qb, 2 | 8);
  gemmH(Xsa, saWk, inF(7), Kb, 2);
  gemmH(Xsa, saWv, inF(9), Vb, 4);
  attn_wmma<<<dim3(T_ / 128, H_, B_), 256, 0, stream>>>(Qb, Kb, Vb, attnO, 1);
  gemmB(attnO, E_, saWp, E_, nullptr, proj, nullptr, E_, NTOK, E_, E_, 0);
  ln_fused<<<NTOK, 256, 0, stream>>>(proj, inF(11), inF(0), inF(24), inF(25), x1f, x1b);

  // ---- cross-attention (full mask)
  gemmH(x1b, caWq, inF(13), Qb, 2 | 8);
  gemmH(Xc,  caWk, inF(15), Kb, 2);
  gemmH(Xc,  caWv, inF(17), Vb, 4);
  attn_wmma<<<dim3(T_ / 128, H_, B_), 256, 0, stream>>>(Qb, Kb, Vb, attnO, 0);
  gemmB(attnO, E_, caWp, E_, nullptr, proj, nullptr, E_, NTOK, E_, E_, 0);
  ln_fused<<<NTOK, 256, 0, stream>>>(proj, inF(19), x1f, inF(26), inF(27), x2f, x2b);

  // ---- FFN
  gemmB(x2b, E_, W1, E_, inF(21), nullptr, hid, FF_, NTOK, FF_, E_, 1);
  gemmB(hid, FF_, W2, FF_, nullptr, ffo, nullptr, E_, NTOK, E_, FF_, 0);
  ln_fused<<<NTOK, 256, 0, stream>>>(ffo, inF(23), x2f, inF(28), inF(29),
                                     (float*)d_out, nullptr);
}
